// LovaszLoss_15805479649596
// MI455X (gfx1250) — matrix operand order repaired and checked by reading, not compiled, not run on hardware
//
#include <hip/hip_runtime.h>
#include <hip/hip_bf16.h>
#include <stdint.h>

#define BB 8
#define CC 19
#define NN 262144              // 512*512
#define SEG (BB*CC)            // 152
#define CHUNK 8                // segments sorted concurrently
#define NCHUNK (SEG/CHUNK)     // 19
#define BLK_ELEMS 4096
#define NBLK (NN/BLK_ELEMS)    // 64

typedef __attribute__((ext_vector_type(16))) _Float16 v16h;
typedef __attribute__((ext_vector_type(8)))  float    v8f;

__device__ __forceinline__ unsigned ballot32(int p) {
  return __builtin_amdgcn_ballot_w32(p != 0);
}

// async global->LDS copy of one 8B record per lane (GV mode), ASYNCcnt-tracked
__device__ __forceinline__ void async_copy_b64(const unsigned long long* gaddr,
                                               unsigned long long* ldsaddr) {
  unsigned lds = (unsigned)(size_t)ldsaddr;                 // LDS byte offset
  unsigned long long ga = (unsigned long long)(size_t)gaddr;
  asm volatile("global_load_async_to_lds_b64 %0, %1, off" :: "v"(lds), "v"(ga) : "memory");
}
#define ASYNC_WAIT(n) asm volatile("s_wait_asynccnt %0" :: "i"(n) : "memory")

// ---------------- init: zero accumulator + per-segment positive counts ------
__global__ void k_init(float* out, unsigned* pcnt) {
  if (threadIdx.x == 0) out[0] = 0.f;
  if (threadIdx.x < SEG) pcnt[threadIdx.x] = 0u;
}

// ---------------- per-(b,c) positive counts ---------------------------------
__global__ void k_pcount(const int* __restrict__ tgt, unsigned* __restrict__ pcnt) {
  int i = blockIdx.x * 256 + threadIdx.x;      // pixel index in [0, B*N)
  int b = i >> 18;
  int t = tgt[i];
  if (t >= 0 && t < CC) atomicAdd(&pcnt[b * CC + t], 1u);
}

// ---------------- per-pixel log-sum-exp (online softmax stats) --------------
__global__ void k_lse(const float* __restrict__ pred, float* __restrict__ lse) {
  int i = blockIdx.x * 256 + threadIdx.x;      // pixel index in [0, B*N)
  int b = i >> 18;
  int n = i & (NN - 1);
  const float* p = pred + (size_t)b * CC * NN + n;
  float mx = -3.402823466e38f, sum = 0.f;
  #pragma unroll
  for (int c = 0; c < CC; ++c) {
    if (c + 1 < CC) __builtin_prefetch(&p[(size_t)(c + 1) * NN], 0, 0); // global_prefetch_b8
    float x  = p[(size_t)c * NN];
    float nm = fmaxf(mx, x);
    sum = sum * __expf(mx - nm) + __expf(x - nm);
    mx  = nm;
  }
  lse[i] = mx + __logf(sum);
}

// ---------------- key generation: (orderable key << 32) | flag --------------
// ascending radix order on key == descending error order; stable radix ==
// stable argsort(-errors), matching the reference's tie behavior.
__global__ void k_keygen(const float* __restrict__ pred, const int* __restrict__ tgt,
                         const float* __restrict__ lse,
                         unsigned long long* __restrict__ out, int segBase) {
  int idx = blockIdx.x * 256 + threadIdx.x;    // [0, CHUNK*N)
  int sl  = idx >> 18;
  int n   = idx & (NN - 1);
  int seg = segBase + sl;
  int b = seg / CC, c = seg % CC;
  float x = pred[(size_t)(b * CC + c) * NN + n];
  float p = __expf(x - lse[b * NN + n]);
  int flag = (tgt[b * NN + n] == c) ? 1 : 0;
  float e  = flag ? (1.f - p) : (1.f + p);
  unsigned bits = __float_as_uint(e);
  unsigned k  = (bits & 0x80000000u) ? ~bits : (bits | 0x80000000u); // ascending-orderable
  unsigned sk = ~k;                                                  // ascend(sk) == descend(e)
  out[(size_t)sl * NN + n] = ((unsigned long long)sk << 32) | (unsigned)flag;
}

// ---------------- radix pass 1/3: per-block 256-bin histograms --------------
__global__ void k_hist(const unsigned long long* __restrict__ recs,
                       unsigned* __restrict__ histg, int shift) {
  __shared__ unsigned long long tile[BLK_ELEMS];   // 32 KB of the 320 KB/WGP LDS
  __shared__ unsigned h[256];
  int sl = blockIdx.x >> 6, blk = blockIdx.x & 63, t = threadIdx.x;
  h[t] = 0;
  const unsigned long long* r = recs + (size_t)sl * NN + blk * BLK_ELEMS;
  for (int k = 0; k < 16; ++k)                      // deep ASYNCcnt pipeline
    async_copy_b64(&r[k * 256 + t], &tile[k * 256 + t]);
  __syncthreads();
  #pragma unroll
  for (int k = 0; k < 16; ++k) {                    // staged drain: in-order completion
    ASYNC_WAIT(15 - k);
    unsigned long long v = tile[k * 256 + t];
    atomicAdd(&h[(unsigned)(v >> shift) & 255u], 1u);
  }
  __syncthreads();
  // layout for the scan: [digit-major, block-minor]
  histg[(size_t)sl * 16384 + t * 64 + blk] = h[t];
}

// ---------------- radix pass 2/3: exclusive scan of 256x64 table ------------
__global__ void k_scan(unsigned* __restrict__ histg) {
  __shared__ unsigned tsum[256];
  unsigned* h = histg + (size_t)blockIdx.x * 16384;
  int t = threadIdx.x;                  // thread t owns digit t's 64 blocks
  unsigned s = 0;
  for (int i = 0; i < 64; ++i) s += h[t * 64 + i];
  tsum[t] = s;
  __syncthreads();
  if (t == 0) {                         // exclusive scan across digits
    unsigned run = 0;
    for (int i = 0; i < 256; ++i) { unsigned v = tsum[i]; tsum[i] = run; run += v; }
  }
  __syncthreads();
  unsigned run = tsum[t];
  for (int i = 0; i < 64; ++i) { unsigned v = h[t * 64 + i]; h[t * 64 + i] = run; run += v; }
}

// ---------------- radix pass 3/3: stable scatter (wave32 ballot ranking) ----
__global__ void k_scatter(const unsigned long long* __restrict__ in,
                          unsigned long long* __restrict__ outp,
                          const unsigned* __restrict__ histg, int shift) {
  __shared__ unsigned long long tile[BLK_ELEMS];   // async-staged input records
  __shared__ unsigned waveCnt[8 * 256];
  __shared__ unsigned runCnt[256];
  int sl = blockIdx.x >> 6, blk = blockIdx.x & 63, t = threadIdx.x;
  int w = t >> 5;
  unsigned lane = t & 31;
  unsigned ltmask = (1u << lane) - 1u;
  runCnt[t] = 0;
  const unsigned long long* rin  = in + (size_t)sl * NN + blk * BLK_ELEMS;
  unsigned long long*       rout = outp + (size_t)sl * NN;
  const unsigned*           hb   = histg + (size_t)sl * 16384;
  for (int k = 0; k < 16; ++k)                      // overlap HBM latency with ranking
    async_copy_b64(&rin[k * 256 + t], &tile[k * 256 + t]);
  __syncthreads();
  #pragma unroll 1
  for (int k = 0; k < 16; ++k) {        // array order = (k, t): stable
    #pragma unroll
    for (int z = 0; z < 8; ++z) waveCnt[z * 256 + t] = 0;
    __syncthreads();
    // each thread consumes only records its own wave issued -> wave-local wait OK
    switch (k) {                        // staged in-order ASYNCcnt drain
      case 0:  ASYNC_WAIT(15); break;  case 1:  ASYNC_WAIT(14); break;
      case 2:  ASYNC_WAIT(13); break;  case 3:  ASYNC_WAIT(12); break;
      case 4:  ASYNC_WAIT(11); break;  case 5:  ASYNC_WAIT(10); break;
      case 6:  ASYNC_WAIT(9);  break;  case 7:  ASYNC_WAIT(8);  break;
      case 8:  ASYNC_WAIT(7);  break;  case 9:  ASYNC_WAIT(6);  break;
      case 10: ASYNC_WAIT(5);  break;  case 11: ASYNC_WAIT(4);  break;
      case 12: ASYNC_WAIT(3);  break;  case 13: ASYNC_WAIT(2);  break;
      case 14: ASYNC_WAIT(1);  break;  default: ASYNC_WAIT(0);  break;
    }
    unsigned long long v = tile[k * 256 + t];
    unsigned d = (unsigned)(v >> shift) & 255u;
    unsigned eq = 0xFFFFFFFFu;          // lanes in my wave with same digit
    #pragma unroll
    for (int bit = 0; bit < 8; ++bit) {
      unsigned bal = ballot32((d >> bit) & 1);
      eq &= ((d >> bit) & 1) ? bal : ~bal;
    }
    unsigned rk = __popc(eq & ltmask);
    if (rk == 0) waveCnt[w * 256 + d] = __popc(eq);   // group leader
    __syncthreads();
    unsigned off = runCnt[d] + rk;
    for (int ww = 0; ww < w; ++ww) off += waveCnt[ww * 256 + d];
    rout[hb[d * 64 + blk] + off] = v;
    __syncthreads();
    unsigned tot = 0;
    #pragma unroll
    for (int ww = 0; ww < 8; ++ww) tot += waveCnt[ww * 256 + t];
    runCnt[t] += tot;
    __syncthreads();
  }
}

// ---------------- per-block positive-flag totals + tiny per-segment scan ----
__global__ void k_fsum(const unsigned long long* __restrict__ recs, unsigned* __restrict__ fsum) {
  __shared__ unsigned long long tile[BLK_ELEMS];
  __shared__ unsigned red[256];
  int sl = blockIdx.x >> 6, blk = blockIdx.x & 63, t = threadIdx.x;
  const unsigned long long* r = recs + (size_t)sl * NN + blk * BLK_ELEMS;
  for (int k = 0; k < 16; ++k)
    async_copy_b64(&r[k * 256 + t], &tile[k * 256 + t]);
  unsigned s = 0;
  #pragma unroll
  for (int k = 0; k < 16; ++k) {
    ASYNC_WAIT(15 - k);
    s += (unsigned)(tile[k * 256 + t] & 1ull);
  }
  red[t] = s;
  __syncthreads();
  for (int o = 128; o > 0; o >>= 1) { if (t < o) red[t] += red[t + o]; __syncthreads(); }
  if (t == 0) fsum[sl * 64 + blk] = red[0];
}

__global__ void k_fscan(unsigned* __restrict__ fsum) {
  __shared__ unsigned v[64];
  int t = threadIdx.x;
  unsigned* f = fsum + blockIdx.x * 64;
  v[t] = f[t];
  __syncthreads();
  if (t == 0) { unsigned run = 0; for (int i = 0; i < 64; ++i) { unsigned x = v[i]; v[i] = run; run += x; } }
  __syncthreads();
  f[t] = v[t];
}

// ---------------- final loss: WMMA triangular-matmul prefix scan + dot ------
__global__ void k_loss(const unsigned long long* __restrict__ recs,
                       const unsigned* __restrict__ fscan,
                       const unsigned* __restrict__ pcnt,
                       float* __restrict__ out, int segBase) {
  __shared__ float ldsTot[256];
  __shared__ float ldsScan[256];
  __shared__ float colOff[16];
  __shared__ float red[256];
  int sl = blockIdx.x >> 6, blk = blockIdx.x & 63, t = threadIdx.x;
  int seg = segBase + sl;
  const unsigned long long* r =
      recs + (size_t)sl * NN + (size_t)blk * BLK_ELEMS + (size_t)t * 16;
  unsigned long long v[16];
  int run = 0;
  #pragma unroll
  for (int j = 0; j < 16; ++j) { v[j] = r[j]; run += (int)(v[j] & 1ull); }
  ldsTot[t] = (float)run;
  __syncthreads();
  // wave 0 computes S = L(16x16 lower-tri ones) * M via v_wmma_f32_16x16x32_f16.
  // M[r][col] = ldsTot[col*16 + r]; counts <=16 are exact in f16, acc in f32.
  if (t < 32) {
    int m = t & 15;
    v16h a = {};
    v16h bm = {};
    #pragma unroll
    for (int e = 0; e < 16; ++e) {            // A rows: K = e (lanes<16) / e+16 (lanes>=16)
      int kk = (t < 16) ? e : (e + 16);
      a[e] = (_Float16)((kk <= m) ? 1.0f : 0.0f);
    }
    if (t < 16) {                             // B rows K=0..15 hold M; K>=16 stay zero
      #pragma unroll
      for (int e = 0; e < 16; ++e) bm[e] = (_Float16)ldsTot[t * 16 + e];
    }
    v8f acc = {};
    acc = __builtin_amdgcn_wmma_f32_16x16x32_f16(false, a, false, bm,
                                                 (short)0, acc, false, false);
    #pragma unroll
    for (int vv = 0; vv < 8; ++vv) {          // D: lane=N(+16), vgpr=M(&7)
      int Mr = (t < 16) ? vv : (vv + 8);
      ldsScan[m * 16 + Mr] = acc[vv];
    }
  }
  __syncthreads();
  if (t == 0) {                               // exclusive scan of 16 column totals
    float runc = 0.f;
    #pragma unroll
    for (int col = 0; col < 16; ++col) { colOff[col] = runc; runc += ldsScan[col * 16 + 15]; }
  }
  __syncthreads();
  float incl = ldsScan[t] + colOff[t >> 4];   // inclusive prefix of thread totals
  int cp = (int)fscan[sl * 64 + blk] + (int)(incl - (float)run);
  float P = (float)pcnt[seg];
  float lsum = 0.f;
  float base = (float)(blk * BLK_ELEMS + t * 16);
  #pragma unroll
  for (int j = 0; j < 16; ++j) {
    int flag = (int)(v[j] & 1ull);
    cp += flag;                               // inclusive positive count at position i
    unsigned k = ~((unsigned)(v[j] >> 32));   // undo orderable transform
    unsigned bits = (k & 0x80000000u) ? (k ^ 0x80000000u) : ~k;
    float e = fmaxf(__uint_as_float(bits), 0.f);
    float c1 = (float)cp;
    float m1 = base + (float)j + 1.f;         // i+1
    float Ja = 1.f - (P - c1) / (P + m1 - c1);
    float m0 = m1 - 1.f;
    float c0 = c1 - (float)flag;
    float Jb = (m0 == 0.f) ? 0.f : (1.f - (P - c0) / (P + m0 - c0));
    lsum += e * (Ja - Jb);
  }
  red[t] = lsum;
  __syncthreads();
  for (int o = 128; o > 0; o >>= 1) { if (t < o) red[t] += red[t + o]; __syncthreads(); }
  if (t == 0) atomicAdd(out, red[0] * (1.0f / (float)SEG));
}

extern "C" void kernel_launch(void* const* d_in, const int* in_sizes, int n_in,
                              void* d_out, int out_size, void* d_ws, size_t ws_size,
                              hipStream_t stream) {
  const float* pred = (const float*)d_in[0];
  const int*   tgt  = (const int*)d_in[1];
  float* out = (float*)d_out;

  char* ws = (char*)d_ws;
  size_t off = 0;
  auto alloc = [&](size_t bytes) -> void* {
    void* p = ws + off;
    off = (off + bytes + 255) & ~(size_t)255;
    return p;
  };
  float*              lse  = (float*)alloc((size_t)BB * NN * sizeof(float));
  unsigned*           pcnt = (unsigned*)alloc(SEG * sizeof(unsigned));
  unsigned long long* recA = (unsigned long long*)alloc((size_t)CHUNK * NN * 8);
  unsigned long long* recB = (unsigned long long*)alloc((size_t)CHUNK * NN * 8);
  unsigned*           hist = (unsigned*)alloc((size_t)CHUNK * 16384 * sizeof(unsigned));
  unsigned*           fsum = (unsigned*)alloc((size_t)CHUNK * NBLK * sizeof(unsigned));
  (void)in_sizes; (void)n_in; (void)out_size; (void)ws_size;

  k_init<<<1, 256, 0, stream>>>(out, pcnt);
  k_pcount<<<(BB * NN) / 256, 256, 0, stream>>>(tgt, pcnt);
  k_lse<<<(BB * NN) / 256, 256, 0, stream>>>(pred, lse);

  const int shifts[4] = {32, 40, 48, 56};
  for (int ch = 0; ch < NCHUNK; ++ch) {
    int segBase = ch * CHUNK;
    k_keygen<<<(CHUNK * NN) / 256, 256, 0, stream>>>(pred, tgt, lse, recA, segBase);
    unsigned long long* bufs[2] = {recA, recB};
    for (int p = 0; p < 4; ++p) {     // LSD radix over the high-32 key: A->B->A->B->A
      unsigned long long* src = bufs[p & 1];
      unsigned long long* dst = bufs[(p & 1) ^ 1];
      k_hist<<<CHUNK * NBLK, 256, 0, stream>>>(src, hist, shifts[p]);
      k_scan<<<CHUNK, 256, 0, stream>>>(hist);
      k_scatter<<<CHUNK * NBLK, 256, 0, stream>>>(src, dst, hist, shifts[p]);
    }
    k_fsum<<<CHUNK * NBLK, 256, 0, stream>>>(recA, fsum);
    k_fscan<<<CHUNK, 64, 0, stream>>>(fsum);
    k_loss<<<CHUNK * NBLK, 256, 0, stream>>>(recA, fsum, pcnt, out, segBase);
  }
}
